// Blocks_43155831390928
// MI455X (gfx1250) — matrix operand order repaired and checked by reading, not compiled, not run on hardware
//
#include <hip/hip_runtime.h>
#include <hip/hip_bf16.h>

// ---------------------------------------------------------------------------
// Fused transformer block for MI455X (gfx1250, wave32, WMMA bf16).
//   B=2048 T=64 C=384 H=6 HS=64 DFF=1536
// One workgroup (8 waves) per batch sequence; all intermediates in LDS.
// Weights prepacked fp32->bf16 into WMMA B-fragment layout in d_ws.
// M-strip-mined GEMMs: each wave owns (ntile x 4 mtiles), so every B fragment
// is loaded once and reused by 4 WMMAs (4x less L2 weight traffic).
// x tile staged into LDS via global_load_async_to_lds_b128 (ASYNCcnt).
// ---------------------------------------------------------------------------

typedef __bf16 bf16_t;
typedef __attribute__((ext_vector_type(16))) __bf16 v16bf;
typedef __attribute__((ext_vector_type(8)))  __bf16 v8bf;
typedef __attribute__((ext_vector_type(8)))  float  v8f;

#define TT   64
#define CC   384
#define NH   6
#define HSZ  64
#define DFFC 1536

__device__ __forceinline__ v8f vzero8() {
  v8f v;
#pragma unroll
  for (int i = 0; i < 8; ++i) v[i] = 0.0f;
  return v;
}

__device__ __forceinline__ v16bf cat8(v8bf lo, v8bf hi) {
  return __builtin_shufflevector(lo, hi, 0, 1, 2, 3, 4, 5, 6, 7,
                                 8, 9, 10, 11, 12, 13, 14, 15);
}

// A fragment (16x32 bf16) from row-major [M][K] LDS tile.
// Lane L: row = m0 + L%16 ; K = k0 + (L>=16 ? 16 : 0) + {0..15} contiguous.
__device__ __forceinline__ v16bf ldA(const bf16_t* p, int lda, int m0, int k0,
                                     int lane) {
  const bf16_t* r = p + (size_t)(m0 + (lane & 15)) * lda + k0 + ((lane >> 4) << 4);
  v8bf lo = *(const v8bf*)(r);
  v8bf hi = *(const v8bf*)(r + 8);
  return cat8(lo, hi);
}

// B fragment (32x16 bf16) from row-major [N][K] LDS tile (i.e. B transposed).
// Lane L: row(N) = n0 + L%16 ; K = k0 + (L>=16 ? 16 : 0) + {0..15} contiguous.
__device__ __forceinline__ v16bf ldBt(const bf16_t* p, int ldk, int n0, int k0,
                                      int lane) {
  const bf16_t* r = p + (size_t)(n0 + (lane & 15)) * ldk + k0 + ((lane >> 4) << 4);
  v8bf lo = *(const v8bf*)(r);
  v8bf hi = *(const v8bf*)(r + 8);
  return cat8(lo, hi);
}

// B fragment from globally prepacked weights: fragment-major, lane-major,
// 16 bf16 (32 bytes) per lane -> one coalesced 1KB read per fragment.
__device__ __forceinline__ v16bf ldBp(const bf16_t* p, int KT, int ntile, int kt,
                                      int lane) {
  const bf16_t* r = p + ((((size_t)ntile * KT) + kt) * 32 + lane) * 16;
  v8bf lo = *(const v8bf*)(r);
  v8bf hi = *(const v8bf*)(r + 8);
  return cat8(lo, hi);
}

__device__ __forceinline__ v8f wmma_bf16(v16bf a, v16bf b, v8f c) {
  return __builtin_amdgcn_wmma_f32_16x16x32_bf16(false, a, false, b, (short)0, c,
                                                 false, false);
}

// Full M-strip (4 m-tiles) for one ntile, B from prepacked global weights.
template <int KT_CNT>
__device__ __forceinline__ void gemm_strip_gB(const bf16_t* A, int lda,
                                              const bf16_t* Bp, int KTtot,
                                              int ntile, int kt0, int lane,
                                              v8f acc[4]) {
  for (int kt = 0; kt < KT_CNT; ++kt) {
    v16bf b = ldBp(Bp, KTtot, ntile, kt0 + kt, lane);
#pragma unroll
    for (int mt = 0; mt < 4; ++mt) {
      v16bf a = ldA(A, lda, mt * 16, kt * 32, lane);
      acc[mt] = wmma_bf16(a, b, acc[mt]);
    }
  }
}

// Full M-strip (4 m-tiles) for one ntile, B transposed in LDS.
template <int KT_CNT>
__device__ __forceinline__ void gemm_strip_lB(const bf16_t* A, int lda,
                                              const bf16_t* Bt, int ldk, int n0,
                                              int lane, v8f acc[4]) {
  for (int kt = 0; kt < KT_CNT; ++kt) {
    v16bf b = ldBt(Bt, ldk, n0, kt * 32, lane);
#pragma unroll
    for (int mt = 0; mt < 4; ++mt) {
      v16bf a = ldA(A, lda, mt * 16, kt * 32, lane);
      acc[mt] = wmma_bf16(a, b, acc[mt]);
    }
  }
}

// LayerNorm over last dim (384), 4 threads per row, wave shuffles for reduce.
__device__ __forceinline__ void ln_rows(const float* src, bf16_t* dst,
                                        const float* w, const float* bb,
                                        int tid) {
  int row = tid >> 2, sub = tid & 3;
  const float* r = src + (size_t)row * CC;
  float s = 0.f, ss = 0.f;
  for (int c = sub; c < CC; c += 4) {
    float v = r[c];
    s += v;
    ss += v * v;
  }
  s  += __shfl_xor(s, 1);  s  += __shfl_xor(s, 2);
  ss += __shfl_xor(ss, 1); ss += __shfl_xor(ss, 2);
  float m   = s * (1.0f / CC);
  float inv = rsqrtf(ss * (1.0f / CC) - m * m + 1e-5f);
  for (int c = sub; c < CC; c += 4)
    dst[row * CC + c] = (bf16_t)((r[c] - m) * inv * w[c] + bb[c]);
}

// ---------------------------------------------------------------------------
// Weight prepack: W viewed as [G][K][Ng] fp32 -> bf16 fragments for B operand.
// dst[((ntile*KT + ktile)*32 + lane)*16 + (k&15)], lane = n%16 | (k bit4)<<4.
// ---------------------------------------------------------------------------
__global__ void pack_b_kernel(const float* __restrict__ src,
                              bf16_t* __restrict__ dst, int K, int N, int Ng) {
  int e = blockIdx.x * blockDim.x + threadIdx.x;
  if (e >= K * N) return;
  int n = e % N;
  int k = e / N;
  int g = n / Ng, ng = n % Ng;
  float v = src[(size_t)g * K * Ng + (size_t)k * Ng + ng];
  int KT = K >> 5;
  int lane = (n & 15) | (((k >> 4) & 1) << 4);
  size_t idx = (((size_t)(n >> 4) * KT + (k >> 5)) * 32 + lane) * 16 + (k & 15);
  dst[idx] = (bf16_t)v;
}

// ---------------------------------------------------------------------------
// Fused block kernel: one workgroup per batch sequence.
// ---------------------------------------------------------------------------
__global__ __launch_bounds__(256) void fused_block(
    const float* __restrict__ x,
    const float* __restrict__ ln1w, const float* __restrict__ ln1b,
    const float* __restrict__ bo,
    const float* __restrict__ ln2w, const float* __restrict__ ln2b,
    const float* __restrict__ b1, const float* __restrict__ b2,
    const bf16_t* __restrict__ pQ, const bf16_t* __restrict__ pK,
    const bf16_t* __restrict__ pV, const bf16_t* __restrict__ pWo,
    const bf16_t* __restrict__ pW1, const bf16_t* __restrict__ pW2,
    float* __restrict__ out) {
  // 288 KB of the 320 KB WGP LDS, phase-overlaid:
  __shared__ __align__(32) unsigned char smem[294912];
  bf16_t* sH  = (bf16_t*)(smem);           // 48K: h(LN1) -> P(softmax) -> h2(LN2)
  bf16_t* sQ  = (bf16_t*)(smem + 49152);   // 48K: Q -> O (concat heads)
  bf16_t* sK  = (bf16_t*)(smem + 98304);   // 48K: K -> MLP hidden chunk
  bf16_t* sVt = (bf16_t*)(smem + 147456);  // 48K: V^T, [384][64]
  float*  sS  = (float*)(smem + 196608);   // 96K: x stage -> S[6][64][64] -> x1

  const int tid  = threadIdx.x;
  const int wave = tid >> 5;
  const int lane = tid & 31;
  const size_t bidx = blockIdx.x;
  const float* xb = x + bidx * (size_t)(TT * CC);

  // ---- Phase A0: async-copy x tile (96KB) into LDS (sS region) ----------
  {
    unsigned ldsbase = (unsigned)(uintptr_t)(void*)sS;
    const char* gb = (const char*)xb;
    for (int c = tid; c < (TT * CC * 4) / 16; c += 256) {
      unsigned la = ldsbase + (unsigned)c * 16u;
      const void* ga = gb + (size_t)c * 16;
      asm volatile("global_load_async_to_lds_b128 %0, %1, off"
                   :: "v"(la), "v"(ga)
                   : "memory");
    }
    asm volatile("s_wait_asynccnt 0x0" ::: "memory");
  }
  __syncthreads();

  // ---- Phase A: LN1(x) -> sH (bf16), reading x from LDS -----------------
  ln_rows(sS, sH, ln1w, ln1b, tid);
  __syncthreads();

  // ---- Phase B: Q,K,V = h @ W{q,k,v}  (3x M=64 N=384 K=384) ------------
  // strips: 3 mats x 24 ntiles = 72, 9 per wave
  for (int t = wave; t < 3 * 24; t += 8) {
    int mat = t / 24, nt = t % 24;
    const bf16_t* Bp = (mat == 0) ? pQ : (mat == 1) ? pK : pV;
    v8f acc[4] = {vzero8(), vzero8(), vzero8(), vzero8()};
    gemm_strip_gB<12>(sH, CC, Bp, 12, nt, 0, lane, acc);
    int n    = nt * 16 + (lane & 15);
    int mofs = (lane >> 4) << 3;
#pragma unroll
    for (int mt = 0; mt < 4; ++mt) {
#pragma unroll
      for (int i = 0; i < 8; ++i) {
        int m = mt * 16 + mofs + i;
        float v = acc[mt][i];
        if (mat == 0)      sQ[m * CC + n] = (bf16_t)v;
        else if (mat == 1) sK[m * CC + n] = (bf16_t)v;
        else               sVt[n * 64 + m] = (bf16_t)v;  // store V transposed
      }
    }
  }
  __syncthreads();

  // ---- Phase C1: S_h = Q_h K_h^T (no 1/sqrt(d) scale, per reference) ----
  // strips: 6 heads x 4 ntiles = 24, 3 per wave
  for (int t = wave; t < NH * 4; t += 8) {
    int h = t / 4, nt = t % 4;
    v8f acc[4] = {vzero8(), vzero8(), vzero8(), vzero8()};
    gemm_strip_lB<2>(sQ + h * HSZ, CC, sK + h * HSZ, CC, nt * 16, lane, acc);
    int n    = nt * 16 + (lane & 15);
    int mofs = (lane >> 4) << 3;
#pragma unroll
    for (int mt = 0; mt < 4; ++mt)
#pragma unroll
      for (int i = 0; i < 8; ++i)
        sS[h * 4096 + (mt * 16 + mofs + i) * 64 + n] = acc[mt][i];
  }
  __syncthreads();

  // ---- Phase C2: causal mask + softmax over QUERY axis t (per column s) -
  bf16_t* sP = sH;  // h consumed; reuse for P
  for (int col = tid; col < NH * 64; col += 256) {
    int h = col >> 6, s = col & 63;
    const float* Sc = sS + h * 4096 + s;  // stride 64 walks t
    float mx = -1e30f;
    for (int tq = s; tq < 64; ++tq) mx = fmaxf(mx, Sc[tq * 64]);
    float sum = 0.f;
    for (int tq = s; tq < 64; ++tq) sum += __expf(Sc[tq * 64] - mx);
    float rinv = 1.0f / sum;
    bf16_t* Pc = sP + h * 4096 + s;
    for (int tq = 0; tq < 64; ++tq) {
      float pv = (tq >= s) ? __expf(Sc[tq * 64] - mx) * rinv : 0.0f;
      Pc[tq * 64] = (bf16_t)pv;
    }
  }
  __syncthreads();

  // ---- Phase C3: O_h = P_h V_h ; concat heads into sQ region ------------
  bf16_t* sO = sQ;  // Q consumed; reuse for O
  for (int t = wave; t < NH * 4; t += 8) {
    int h = t / 4, nt = t % 4;
    v8f acc[4] = {vzero8(), vzero8(), vzero8(), vzero8()};
    gemm_strip_lB<2>(sP + h * 4096, 64, sVt + h * HSZ * 64, 64, nt * 16, lane,
                     acc);
    int n    = nt * 16 + (lane & 15);
    int mofs = (lane >> 4) << 3;
#pragma unroll
    for (int mt = 0; mt < 4; ++mt)
#pragma unroll
      for (int i = 0; i < 8; ++i)
        sO[(mt * 16 + mofs + i) * CC + h * HSZ + n] = (bf16_t)acc[mt][i];
  }
  __syncthreads();

  // ---- Phase D: x1 = x + O @ Wo + bo  (f32, into sS region) -------------
  float* sX1 = sS;
  for (int t = wave; t < 24; t += 8) {
    int nt = t;
    v8f acc[4] = {vzero8(), vzero8(), vzero8(), vzero8()};
    gemm_strip_gB<12>(sO, CC, pWo, 12, nt, 0, lane, acc);
    int n    = nt * 16 + (lane & 15);
    int mofs = (lane >> 4) << 3;
    float bn = bo[n];
#pragma unroll
    for (int mt = 0; mt < 4; ++mt)
#pragma unroll
      for (int i = 0; i < 8; ++i) {
        int m = mt * 16 + mofs + i;
        sX1[m * CC + n] = acc[mt][i] + bn + xb[m * CC + n];  // residual (L2 hit)
      }
  }
  __syncthreads();

  // ---- Phase E: h2 = LN2(x1) -> sH ------------------------------------
  ln_rows(sX1, sH, ln2w, ln2b, tid);
  __syncthreads();

  // ---- Phase F: x1 += relu(h2 @ W1 + b1) @ W2, DFF in 12 chunks of 128 --
  bf16_t* sHid = sK;  // K consumed; reuse for hidden chunk [64][128]
  for (int ch = 0; ch < 12; ++ch) {
    // GEMM1: M=64 N=128 K=384, fused bias + ReLU (8 strips, 1 per wave)
    for (int t = wave; t < 8; t += 8) {
      int nt = t;
      v8f acc[4] = {vzero8(), vzero8(), vzero8(), vzero8()};
      gemm_strip_gB<12>(sH, CC, pW1, 12, ch * 8 + nt, 0, lane, acc);
      int n    = nt * 16 + (lane & 15);
      int mofs = (lane >> 4) << 3;
      float bn = b1[ch * 128 + n];
#pragma unroll
      for (int mt = 0; mt < 4; ++mt)
#pragma unroll
        for (int i = 0; i < 8; ++i)
          sHid[(mt * 16 + mofs + i) * 128 + n] =
              (bf16_t)fmaxf(acc[mt][i] + bn, 0.0f);
    }
    __syncthreads();
    // GEMM2: M=64 N=384 K=128, accumulate into x1 (3 strips per wave)
    for (int t = wave; t < 24; t += 8) {
      int nt = t;
      v8f acc[4] = {vzero8(), vzero8(), vzero8(), vzero8()};
      gemm_strip_gB<4>(sHid, 128, pW2, 48, nt, ch * 4, lane, acc);
      int n    = nt * 16 + (lane & 15);
      int mofs = (lane >> 4) << 3;
#pragma unroll
      for (int mt = 0; mt < 4; ++mt)
#pragma unroll
        for (int i = 0; i < 8; ++i)
          sX1[(mt * 16 + mofs + i) * CC + n] += acc[mt][i];
    }
    __syncthreads();
  }

  // ---- Phase G: out = x1 + b2 ------------------------------------------
  float* ob = out + bidx * (size_t)(TT * CC);
  for (int idx = tid; idx < TT * CC; idx += 256)
    ob[idx] = sX1[idx] + b2[idx % CC];
}

// ---------------------------------------------------------------------------
extern "C" void kernel_launch(void* const* d_in, const int* in_sizes, int n_in,
                              void* d_out, int out_size, void* d_ws,
                              size_t ws_size, hipStream_t stream) {
  (void)in_sizes; (void)n_in; (void)out_size; (void)ws_size;

  const float* x    = (const float*)d_in[0];
  const float* ln1w = (const float*)d_in[1];
  const float* ln1b = (const float*)d_in[2];
  const float* Wq   = (const float*)d_in[3];
  const float* Wk   = (const float*)d_in[4];
  const float* Wv   = (const float*)d_in[5];
  const float* Wo   = (const float*)d_in[6];
  const float* bo   = (const float*)d_in[7];
  const float* ln2w = (const float*)d_in[8];
  const float* ln2b = (const float*)d_in[9];
  const float* W1   = (const float*)d_in[10];
  const float* b1   = (const float*)d_in[11];
  const float* W2   = (const float*)d_in[12];
  const float* b2   = (const float*)d_in[13];
  float* out = (float*)d_out;

  // Workspace: prepacked bf16 weights (~3.4 MB)
  bf16_t* ws  = (bf16_t*)d_ws;
  bf16_t* pQ  = ws;
  bf16_t* pK  = pQ + 384 * 384;
  bf16_t* pV  = pK + 384 * 384;
  bf16_t* pWo = pV + 384 * 384;
  bf16_t* pW1 = pWo + 384 * 384;
  bf16_t* pW2 = pW1 + 384 * 1536;

  const int thr = 256;
  pack_b_kernel<<<(384 * 384 + thr - 1) / thr, thr, 0, stream>>>(Wq, pQ, 384, 384, 64);
  pack_b_kernel<<<(384 * 384 + thr - 1) / thr, thr, 0, stream>>>(Wk, pK, 384, 384, 64);
  pack_b_kernel<<<(384 * 384 + thr - 1) / thr, thr, 0, stream>>>(Wv, pV, 384, 384, 64);
  pack_b_kernel<<<(384 * 384 + thr - 1) / thr, thr, 0, stream>>>(Wo, pWo, 384, 384, 384);
  pack_b_kernel<<<(384 * 1536 + thr - 1) / thr, thr, 0, stream>>>(W1, pW1, 384, 1536, 1536);
  pack_b_kernel<<<(1536 * 384 + thr - 1) / thr, thr, 0, stream>>>(W2, pW2, 1536, 384, 384);

  fused_block<<<2048, 256, 0, stream>>>(x, ln1w, ln1b, bo, ln2w, ln2b, b1, b2,
                                        pQ, pK, pV, pWo, pW1, pW2, out);
}
